// KANLinear_2946347565196
// MI455X (gfx1250) — matrix operand order repaired and checked by reading, not compiled, not run on hardware
//
#include <hip/hip_runtime.h>

// ---------------------------------------------------------------------------
// KAN linear layer on gfx1250:
//   out(4096,1024) = silu(x) @ base_w^T + einsum('bik,oik', bases, sw*scaler)
// Fused as one bf16 WMMA GEMM:  A(4096x9216) @ W(1024x9216)^T, f32 accum.
// K layout: [0..1023] = base part (silu / base_weight),
//           [1024 + i*8 + j]  = spline part (basis_j(x_i) / sw[o][i][j]*sc).
// Tile staging uses the CDNA5 async global->LDS pipe (ASYNCcnt), bypassing
// VGPRs entirely; WMMA overlaps the DMA of the next K-slice.
// ---------------------------------------------------------------------------

typedef __bf16 v8bf  __attribute__((ext_vector_type(8)));
typedef __bf16 v16bf __attribute__((ext_vector_type(16)));
typedef float  v8f   __attribute__((ext_vector_type(8)));

#define IN_F   1024
#define OUT_F  1024
#define BATCH  4096
#define KTOT   9216          // 1024 + 1024*8
#define NKNOT  12            // grid_size + 2*order + 1
#define NCOEF  8             // grid_size + order
#define EPSV   1e-8f

// ---------------------------------------------------------------------------
// Build augmented activation matrix A (bf16), row-major (BATCH x KTOT).
// ---------------------------------------------------------------------------
__global__ __launch_bounds__(256) void kan_build_A(
    const float* __restrict__ x, const float* __restrict__ grid,
    __bf16* __restrict__ A)
{
    int idx = blockIdx.x * 256 + threadIdx.x;      // 0 .. BATCH*IN_F-1
    int b = idx >> 10;
    int i = idx & (IN_F - 1);

    float xv = x[idx];
    float sil = xv / (1.0f + __expf(-xv));         // silu

    float g[NKNOT];
#pragma unroll
    for (int t = 0; t < NKNOT; ++t) g[t] = grid[i * NKNOT + t];

    // Cox-de Boor, degree 3
    float bas[NKNOT - 1];
#pragma unroll
    for (int t = 0; t < NKNOT - 1; ++t)
        bas[t] = (xv >= g[t] && xv < g[t + 1]) ? 1.0f : 0.0f;
#pragma unroll
    for (int k = 1; k <= 3; ++k) {
#pragma unroll
        for (int t = 0; t < NKNOT - 1 - k; ++t) {
            float left  = (xv - g[t]) / (g[t + k] - g[t] + EPSV) * bas[t];
            float right = (g[t + k + 1] - xv) / (g[t + k + 1] - g[t + 1] + EPSV) * bas[t + 1];
            bas[t] = left + right;
        }
    }

    size_t ro = (size_t)b * KTOT;
    A[ro + i] = (__bf16)sil;
#pragma unroll
    for (int j = 0; j < NCOEF; ++j)
        A[ro + IN_F + i * NCOEF + j] = (__bf16)bas[j];
}

// ---------------------------------------------------------------------------
// Build augmented weight matrix W (bf16), row-major (OUT_F x KTOT).
// ---------------------------------------------------------------------------
__global__ __launch_bounds__(256) void kan_build_W(
    const float* __restrict__ bw, const float* __restrict__ sw,
    const float* __restrict__ sc, __bf16* __restrict__ W)
{
    int idx = blockIdx.x * 256 + threadIdx.x;      // 0 .. OUT_F*IN_F-1
    int i = idx & (IN_F - 1);

    float scale = sc[idx];
    size_t ro = (size_t)(idx >> 10) * KTOT;
    W[ro + i] = (__bf16)bw[idx];
#pragma unroll
    for (int j = 0; j < NCOEF; ++j)
        W[ro + IN_F + i * NCOEF + j] = (__bf16)(sw[(size_t)idx * NCOEF + j] * scale);
}

// ---------------------------------------------------------------------------
// CDNA5 async-copy helpers (portable across ROCm7.2 / amdgpu-toolchain:
// inline asm bypasses the arity-divergent TDM builtins).
// The instruction's offset immediate applies to BOTH the LDS and global
// address, so one address pair covers a 32 B per-lane slice in two b128 ops.
// ---------------------------------------------------------------------------
__device__ __forceinline__ unsigned lds_addr32(const void* p)
{
    // LDS aperture keeps the byte offset in addr[31:0]; async-LDS ops add
    // the wave's LDS_BASE in hardware.
    return (unsigned)(size_t)p;
}

__device__ __forceinline__ void async_copy32B(unsigned lds, const __bf16* g)
{
    asm volatile(
        "global_load_async_to_lds_b128 %0, %1, off\n\t"
        "global_load_async_to_lds_b128 %0, %1, off offset:16"
        :: "v"(lds), "v"(g) : "memory");
}

__device__ __forceinline__ void wait_async_all()
{
    asm volatile("s_wait_asynccnt 0x0" ::: "memory");
}

// ---------------------------------------------------------------------------
// GEMM: C(M,N) = A(M,K) * W(N,K)^T  with v_wmma_f32_16x16x32_bf16.
// Block = 256 threads (8 waves), 128x128 C tile, BK=32, double-buffered LDS
// filled by the async global->LDS pipe. Wave grid 4 (M) x 2 (N); each wave
// owns 2x4 16x16 f32 accumulators (8 WMMAs per K-step).
// LDS rows padded to 40 elements (80 B) -> 16B-aligned b128, bank-spread.
// ---------------------------------------------------------------------------
#define BM 128
#define BN 128
#define BK 32
#define LDK 40   // padded LDS row length in bf16 elements (80 bytes)

__global__ __launch_bounds__(256) void kan_gemm(
    const __bf16* __restrict__ A, const __bf16* __restrict__ W,
    float* __restrict__ C)
{
    __shared__ __bf16 sA[2][BM * LDK];
    __shared__ __bf16 sB[2][BN * LDK];

    const int tid  = threadIdx.x;
    const int bm   = blockIdx.x >> 3;      // 32 M-blocks
    const int bn   = blockIdx.x & 7;       // 8 N-blocks
    const int m0   = bm * BM;
    const int n0   = bn * BN;
    const int lane = tid & 31;
    const int wave = tid >> 5;
    const int wm   = wave & 3;             // wave M position (0..3)
    const int wn   = wave >> 2;            // wave N position (0..1)
    const int fl   = lane & 15;            // fragment lane row/col
    const int fh   = lane >> 4;            // fragment half select

    // Staging map: thread t owns a 32 B slice: row = t>>1, half = t&1.
    const int lrow  = tid >> 1;
    const int lhalf = tid & 1;
    const __bf16* gA = A + (size_t)(m0 + lrow) * KTOT + lhalf * 16;
    const __bf16* gB = W + (size_t)(n0 + lrow) * KTOT + lhalf * 16;
    unsigned ldsA[2], ldsB[2];
    ldsA[0] = lds_addr32(&sA[0][lrow * LDK + lhalf * 16]);
    ldsA[1] = lds_addr32(&sA[1][lrow * LDK + lhalf * 16]);
    ldsB[0] = lds_addr32(&sB[0][lrow * LDK + lhalf * 16]);
    ldsB[1] = lds_addr32(&sB[1][lrow * LDK + lhalf * 16]);

    v8f acc[2][4] = {};
    const int nk = KTOT / BK;              // 288 K-steps

    // --- prologue: DMA tile 0 into buffer 0 ---
    async_copy32B(ldsA[0], gA);
    async_copy32B(ldsB[0], gB);
    wait_async_all();
    __syncthreads();

    for (int kt = 0; kt < nk; ++kt) {
        const int cur = kt & 1;
        const int nxt = cur ^ 1;

        // Kick off DMA of the next K-slice into the other buffer; it runs
        // concurrently with the WMMAs below (ASYNCcnt tracks completion).
        if (kt + 1 < nk) {
            const size_t ko = (size_t)(kt + 1) * BK;     // element offset
            async_copy32B(ldsA[nxt], gA + ko);
            async_copy32B(ldsB[nxt], gB + ko);
        }

        // Load WMMA fragments from LDS (layouts match ISA 16-bit striping).
        v16bf afrag[2], bfrag[4];
#pragma unroll
        for (int mt = 0; mt < 2; ++mt) {
            const __bf16* base = &sA[cur][(wm * 32 + mt * 16 + fl) * LDK];
            // lane 0-15: K 0-7 & 16-23 of row fl; lane 16-31: K 8-15 & 24-31.
            v8bf lo = *(const v8bf*)(base + fh * 8);
            v8bf hi = *(const v8bf*)(base + 16 + fh * 8);
            afrag[mt] = __builtin_shufflevector(lo, hi,
                0, 1, 2, 3, 4, 5, 6, 7, 8, 9, 10, 11, 12, 13, 14, 15);
        }
#pragma unroll
        for (int nt = 0; nt < 4; ++nt) {
            // lane 0-15: col fl, K 0-7 & 16-23; lane 16-31: col fl, K 8-15 & 24-31.
            const __bf16* base = &sB[cur][(wn * 64 + nt * 16 + fl) * LDK];
            v8bf lo = *(const v8bf*)(base + fh * 8);
            v8bf hi = *(const v8bf*)(base + 16 + fh * 8);
            bfrag[nt] = __builtin_shufflevector(lo, hi,
                0, 1, 2, 3, 4, 5, 6, 7, 8, 9, 10, 11, 12, 13, 14, 15);
        }

        // 8 WMMAs per wave per K-step.
#pragma unroll
        for (int mt = 0; mt < 2; ++mt)
#pragma unroll
            for (int nt = 0; nt < 4; ++nt)
                acc[mt][nt] = __builtin_amdgcn_wmma_f32_16x16x32_bf16(
                    false, afrag[mt], false, bfrag[nt],
                    (short)0, acc[mt][nt], false, false);

        // Pipeline boundary: our async writes done, then all waves' writes
        // visible (barrier) before the next iteration reads buffer nxt.
        if (kt + 1 < nk) {
            wait_async_all();
            __syncthreads();
        }
    }

    // Epilogue: C/D layout — VGPR r, lanes 0-15: M=r, N=lane;
    //                              lanes 16-31: M=8+r, N=lane-16.
    const int crow0 = m0 + wm * 32;
    const int ccol0 = n0 + wn * 64;
#pragma unroll
    for (int mt = 0; mt < 2; ++mt) {
#pragma unroll
        for (int nt = 0; nt < 4; ++nt) {
            const v8f a = acc[mt][nt];
            const int col   = ccol0 + nt * 16 + fl;
            const int rbase = crow0 + mt * 16 + fh * 8;
#pragma unroll
            for (int r = 0; r < 8; ++r)
                C[(size_t)(rbase + r) * OUT_F + col] = a[r];
        }
    }
}

// ---------------------------------------------------------------------------
extern "C" void kernel_launch(void* const* d_in, const int* in_sizes, int n_in,
                              void* d_out, int out_size, void* d_ws, size_t ws_size,
                              hipStream_t stream)
{
    const float* x    = (const float*)d_in[0];  // (4096, 1024)
    const float* bw   = (const float*)d_in[1];  // (1024, 1024)
    const float* sw   = (const float*)d_in[2];  // (1024, 1024, 8)
    const float* sc   = (const float*)d_in[3];  // (1024, 1024)
    const float* grid = (const float*)d_in[4];  // (1024, 12)

    __bf16* Aw = (__bf16*)d_ws;                                     // 75.5 MB
    __bf16* Ww = (__bf16*)((char*)d_ws + (size_t)BATCH * KTOT * 2); // 18.9 MB

    kan_build_A<<<(BATCH * IN_F) / 256, 256, 0, stream>>>(x, grid, Aw);
    kan_build_W<<<(OUT_F * IN_F) / 256, 256, 0, stream>>>(bw, sw, sc, Ww);
    kan_gemm<<<(BATCH / BM) * (OUT_F / BN), 256, 0, stream>>>(Aw, Ww, (float*)d_out);
}
